// LogicMachine_48309792145457
// MI455X (gfx1250) — matrix operand order repaired and checked by reading, not compiled, not run on hardware
//
#include <hip/hip_runtime.h>
#include <hip/hip_bf16.h>
#include <math.h>

typedef __attribute__((ext_vector_type(16))) _Float16 v16h;
typedef __attribute__((ext_vector_type(8)))  float    v8f;

// ---------------------------------------------------------------------------
// Problem constants (from reference): B=8, n=32, C=64, O=64
// ---------------------------------------------------------------------------
#define NB 8
#define NN 32
#define NC 64
#define NO 64

// f16 weight-fragment buffer layout (offsets in halves, each matrix stored in
// WMMA-B-fragment order: frag(ks,nt) -> 32 lanes x 16 halves contiguous)
#define WOFF_OP0   0        // K=64
#define WOFF_RED0  4096     // K=128
#define WOFF_EXP1  12288    // K=64
#define WOFF_OP1   16384    // K=64
#define WOFF_RED1  20480    // K=128
#define WOFF_EXP2  28672    // K=128
#define WOFF_OP2   36864    // K=128
#define WOFF_RED2  45056    // K=256
#define WOFF_EXP3  61440    // K=384
#define WOFF_OP3   86016    // K=384
#define WHALF_TOTAL 110592

// workspace layout (bytes)
#define WS_REDA_OFF   (WHALF_TOTAL * 2)                        // 221184
#define WS_REDB_OFF   (WS_REDA_OFF + NB*NN*NN*2*NC*4)          // + 4 MB
#define WS_REDC_OFF   (WS_REDB_OFF + NB*NN*2*NC*4)
#define WS_BASE_END   (WS_REDC_OFF + NB*2*NC*4)                // 4550656
#define WS_X3H_OFF    WS_BASE_END                              // f16 x3 copy
#define WS_X3H_BYTES  ((size_t)NB*NN*NN*NN*NC*2)               // 33554432
#define WS_X2H_OFF    (WS_X3H_OFF + WS_X3H_BYTES)
#define WS_X2H_BYTES  ((size_t)NB*NN*NN*NC*2)                  // 1048576
#define WS_F16_END    (WS_X2H_OFF + WS_X2H_BYTES)              // ~39.2 MB

// ---------------------------------------------------------------------------
// Weight prep: fp32 [O, K] row-major  ->  f16 WMMA B-fragment order.
// For V_WMMA_F32_16X16X32_F16, B (32x16, f16) fragment per lane:
//   n = nt*16 + (lane & 15); k = ks*32 + (lane<16 ? 0 : 8) + (h&7) + (h>=8 ? 16 : 0), h in [0,16)
// ---------------------------------------------------------------------------
__global__ void prep_w_kernel(const float* __restrict__ w,
                              _Float16* __restrict__ dst,
                              int Ktot, int total) {
    int idx = blockIdx.x * blockDim.x + threadIdx.x;
    if (idx >= total) return;
    int h  = idx & 15;
    int l  = (idx >> 4) & 31;
    int f  = idx >> 9;
    int nt = f & 3;
    int ks = f >> 2;
    int nn = nt * 16 + (l & 15);
    int k  = ks * 32 + ((l < 16) ? 0 : 8) + (h & 7) + ((h & 8) << 1);
    dst[idx] = (_Float16)w[nn * Ktot + k];
}

// fp32 -> f16 bulk conversion (4 elements / thread, totals are multiples of 4)
__global__ void conv_f16_kernel(const float* __restrict__ src,
                                _Float16* __restrict__ dst, int total4) {
    int t = blockIdx.x * blockDim.x + threadIdx.x;
    if (t >= total4) return;
    int idx = t * 4;
    float4 v = *(const float4*)(src + idx);
    dst[idx + 0] = (_Float16)v.x;
    dst[idx + 1] = (_Float16)v.y;
    dst[idx + 2] = (_Float16)v.z;
    dst[idx + 3] = (_Float16)v.w;
}

// ---------------------------------------------------------------------------
// Reducers: masked exists(max)/forall(min) over last object axis, channels
// interleaved -> 2C. Single pass over the big tensor (bandwidth-bound).
// ---------------------------------------------------------------------------
__global__ void reduceA_kernel(const float* __restrict__ x3,
                               float* __restrict__ redA) {
    int idx = blockIdx.x * blockDim.x + threadIdx.x;   // NB*NN*NN*NC threads
    int c = idx & (NC - 1);
    int r = idx >> 6;
    int j = r & (NN - 1);
    int i = (r >> 5) & (NN - 1);
    int b = r >> 10;
    const float* src = x3 + ((size_t)(((b * NN + i) * NN + j)) * NN) * NC + c;
    float ex = -3.4e38f, fa = 3.4e38f;
    bool ij = (i != j);
    for (int k = 0; k < NN; ++k) {
        float xv = src[(size_t)k * NC];
        bool mk = ij && (k != i) && (k != j);
        ex = fmaxf(ex, mk ? xv : 0.0f);
        fa = fminf(fa, mk ? xv : 1.0f);
    }
    float* d = redA + ((size_t)((b * NN + i) * NN + j)) * (2 * NC);
    d[2 * c] = ex;
    d[2 * c + 1] = fa;
}

__global__ void reduceB_kernel(const float* __restrict__ x2,
                               float* __restrict__ redB) {
    int idx = blockIdx.x * blockDim.x + threadIdx.x;   // NB*NN*NC threads
    int c = idx & (NC - 1);
    int r = idx >> 6;
    int i = r & (NN - 1);
    int b = r >> 5;
    const float* src = x2 + ((size_t)((b * NN + i) * NN)) * NC + c;
    float ex = -3.4e38f, fa = 3.4e38f;
    for (int j = 0; j < NN; ++j) {
        float xv = src[(size_t)j * NC];
        bool mk = (i != j);
        ex = fmaxf(ex, mk ? xv : 0.0f);
        fa = fminf(fa, mk ? xv : 1.0f);
    }
    float* d = redB + ((size_t)(b * NN + i)) * (2 * NC);
    d[2 * c] = ex;
    d[2 * c + 1] = fa;
}

__global__ void reduceC_kernel(const float* __restrict__ x1,
                               float* __restrict__ redC) {
    int idx = blockIdx.x * blockDim.x + threadIdx.x;   // NB*NC threads
    if (idx >= NB * NC) return;
    int c = idx & (NC - 1);
    int b = idx >> 6;
    const float* src = x1 + ((size_t)(b * NN)) * NC + c;
    float ex = -3.4e38f, fa = 3.4e38f;
    for (int k = 0; k < NN; ++k) {
        float xv = src[(size_t)k * NC];
        ex = fmaxf(ex, xv);
        fa = fminf(fa, xv);
    }
    float* d = redC + (size_t)b * (2 * NC);
    d[2 * c] = ex;
    d[2 * c + 1] = fa;
}

// ---------------------------------------------------------------------------
// WMMA helpers
// A (16x32, f16) fragment per lane (documented layout):
//   lanes 0-15 : M=lane,    halves 0-7 -> K=c0..c0+7,  8-15 -> K=c0+16..c0+23 (c0=kb*32)
//   lanes 16-31: M=lane-16, same with c0 = kb*32+8
// ---------------------------------------------------------------------------
__device__ __forceinline__ v8f wmma_acc(v8f acc, v16h a, v16h b) {
    return __builtin_amdgcn_wmma_f32_16x16x32_f16(
        /*neg_a=*/false, a, /*neg_b=*/false, b,
        /*c_mod=*/(short)0, acc, /*reuse_a=*/false, /*reuse_b=*/false);
}

template <typename T>
__device__ __forceinline__ v16h load_afrag(const T* __restrict__ rp, int c0) {
    v16h a;
#pragma unroll
    for (int h = 0; h < 8; ++h) {
        a[h]     = (_Float16)rp[c0 + h];
        a[h + 8] = (_Float16)rp[c0 + 16 + h];
    }
    return a;
}

__device__ __forceinline__ v16h load_afrag_s(const float* __restrict__ rp, int c0, float s) {
    v16h a;
#pragma unroll
    for (int h = 0; h < 8; ++h) {
        a[h]     = (_Float16)(rp[c0 + h] * s);
        a[h + 8] = (_Float16)(rp[c0 + 16 + h] * s);
    }
    return a;
}

__device__ __forceinline__ v16h load_bfrag(const _Float16* __restrict__ wf,
                                           int ks, int nt, int lane) {
    return *(const v16h*)(wf + ((((ks << 2) + nt) << 5) + lane) * 16);
}

// One 64-feature K-chunk (2 x K=32 WMMA steps across 4 N-tiles)
template <typename T>
__device__ __forceinline__ void mm_chunk(v8f (&acc)[4], const T* __restrict__ rp,
                                         const _Float16* __restrict__ wf,
                                         int chunk, int lane) {
#pragma unroll
    for (int kb = 0; kb < 2; ++kb) {
        v16h a = load_afrag(rp, kb * 32 + ((lane < 16) ? 0 : 8));
        int ks = chunk * 2 + kb;
#pragma unroll
        for (int nt = 0; nt < 4; ++nt)
            acc[nt] = wmma_acc(acc[nt], a, load_bfrag(wf, ks, nt, lane));
    }
}

__device__ __forceinline__ void mm_chunk_s(v8f (&acc)[4], const float* __restrict__ rp,
                                           const _Float16* __restrict__ wf,
                                           int chunk, int lane, float s) {
#pragma unroll
    for (int kb = 0; kb < 2; ++kb) {
        v16h a = load_afrag_s(rp, kb * 32 + ((lane < 16) ? 0 : 8), s);
        int ks = chunk * 2 + kb;
#pragma unroll
        for (int nt = 0; nt < 4; ++nt)
            acc[nt] = wmma_acc(acc[nt], a, load_bfrag(wf, ks, nt, lane));
    }
}

__device__ __forceinline__ float sigmoidf_(float v) {
    return 1.0f / (1.0f + __expf(-v));
}

// ---------------------------------------------------------------------------
// Order 3 (dominant): one wave per 16-row tile (fixed b,i,j; k = kt*16+m).
// s3 = m8 * Lin(permcat(expand(x2)), w_exp3) + m9 * Lin(permcat(x3), w_op3)
// Templated on activation element type: f16 path reads pre-converted copies
// (half the bytes, no inner-loop cvt; x3h is L2-resident at 33.5 MB).
// ---------------------------------------------------------------------------
template <typename T>
__global__ void gemm3_kernel(const T* __restrict__ x2e, const T* __restrict__ x3e,
                             const int* __restrict__ act,
                             const _Float16* __restrict__ wf_exp3,
                             const _Float16* __restrict__ wf_op3,
                             const float* __restrict__ b_exp3,
                             const float* __restrict__ b_op3,
                             float* __restrict__ o3) {
    int gid  = blockIdx.x * blockDim.x + threadIdx.x;
    int wave = gid >> 5;
    int lane = threadIdx.x & 31;
    int kt = wave & 1; int rest = wave >> 1;
    int j = rest & (NN - 1); rest >>= 5;
    int i = rest & (NN - 1);
    int b = rest >> 5;
    int m = lane & 15;
    int w = kt * 16 + m;

    // 6 permutations: source (i1,i2,i3) for target (u,v,w)=(i,j,w)
    int s1[6]  = { i, i, j, w, j, w };
    int s2[6]  = { j, w, i, i, w, j };
    int s3a[6] = { w, j, w, j, i, i };

    int m8 = act[b * 10 + 8];
    int m9 = act[b * 10 + 9];
    int gate = 0;
#pragma unroll
    for (int bb = 0; bb < NB; ++bb) gate |= act[bb * 10 + 8] | act[bb * 10 + 9];

    v8f z = {0.f, 0.f, 0.f, 0.f, 0.f, 0.f, 0.f, 0.f};
    v8f acc[4] = { z, z, z, z };

    if (m8) {   // expand branch: rows come from x2[b, i1, i2, :]
#pragma unroll
        for (int p = 0; p < 6; ++p) {
            const T* rp = x2e + ((size_t)((b * NN + s1[p]) * NN + s2[p])) * NC;
            mm_chunk(acc, rp, wf_exp3, p, lane);
        }
    }
    if (m9) {   // direct branch: rows from x3[b, i1, i2, i3, :]
#pragma unroll
        for (int p = 0; p < 6; ++p) {
            const T* rp = x3e + ((size_t)(((b * NN + s1[p]) * NN + s2[p]) * NN + s3a[p])) * NC;
            mm_chunk(acc, rp, wf_op3, p, lane);
        }
    }

    size_t base = (size_t)(((b * NN + i) * NN + j) * NN + kt * 16);
    int ncol = lane & 15;
    float fm8 = (float)m8, fm9 = (float)m9;
#pragma unroll
    for (int nt = 0; nt < 4; ++nt) {
        int col = nt * 16 + ncol;
        float bc = fm8 * b_exp3[col] + fm9 * b_op3[col];
#pragma unroll
        for (int r = 0; r < 8; ++r) {
            int mrow = r + ((lane < 16) ? 0 : 8);
            float v = acc[nt][r] + bc;
            o3[(base + mrow) * NO + col] = gate ? sigmoidf_(v) : 0.0f;
        }
    }
}

// ---------------------------------------------------------------------------
// Order 2: tile = fixed (b,i), j = jt*16+m.  Branches bits 5/6/7.
// ---------------------------------------------------------------------------
__global__ void gemm2_kernel(const float* __restrict__ x1, const float* __restrict__ x2,
                             const float* __restrict__ redA,
                             const int* __restrict__ act,
                             const _Float16* __restrict__ wf_exp2,
                             const _Float16* __restrict__ wf_op2,
                             const _Float16* __restrict__ wf_red2,
                             const float* __restrict__ b_exp2,
                             const float* __restrict__ b_op2,
                             const float* __restrict__ b_red2,
                             float* __restrict__ o2) {
    int gid  = blockIdx.x * blockDim.x + threadIdx.x;
    int wave = gid >> 5;
    int lane = threadIdx.x & 31;
    int jt = wave & 1; int rest = wave >> 1;
    int i = rest & (NN - 1);
    int b = rest >> 5;
    int m = lane & 15;
    int j = jt * 16 + m;

    int m5 = act[b * 10 + 5], m6 = act[b * 10 + 6], m7 = act[b * 10 + 7];
    int gate = 0;
#pragma unroll
    for (int bb = 0; bb < NB; ++bb)
        gate |= act[bb * 10 + 5] | act[bb * 10 + 6] | act[bb * 10 + 7];

    v8f z = {0.f, 0.f, 0.f, 0.f, 0.f, 0.f, 0.f, 0.f};
    v8f acc[4] = { z, z, z, z };

    if (m5) {   // permcat(expand(x1)): [x1[b,i], x1[b,j]]
        mm_chunk(acc, x1 + ((size_t)(b * NN + i)) * NC, wf_exp2, 0, lane);
        mm_chunk(acc, x1 + ((size_t)(b * NN + j)) * NC, wf_exp2, 1, lane);
    }
    if (m6) {   // permcat(x2): [x2[b,i,j], x2[b,j,i]]
        mm_chunk(acc, x2 + ((size_t)((b * NN + i) * NN + j)) * NC, wf_op2, 0, lane);
        mm_chunk(acc, x2 + ((size_t)((b * NN + j) * NN + i)) * NC, wf_op2, 1, lane);
    }
    if (m7) {   // permcat(reduce(x3,3)): [redA[b,i,j,:128], redA[b,j,i,:128]]
        const float* r0 = redA + ((size_t)((b * NN + i) * NN + j)) * (2 * NC);
        const float* r1 = redA + ((size_t)((b * NN + j) * NN + i)) * (2 * NC);
        mm_chunk(acc, r0,      wf_red2, 0, lane);
        mm_chunk(acc, r0 + NC, wf_red2, 1, lane);
        mm_chunk(acc, r1,      wf_red2, 2, lane);
        mm_chunk(acc, r1 + NC, wf_red2, 3, lane);
    }

    size_t base = (size_t)((b * NN + i) * NN + jt * 16);
    int ncol = lane & 15;
    float fm5 = (float)m5, fm6 = (float)m6, fm7 = (float)m7;
#pragma unroll
    for (int nt = 0; nt < 4; ++nt) {
        int col = nt * 16 + ncol;
        float bc = fm5 * b_exp2[col] + fm6 * b_op2[col] + fm7 * b_red2[col];
#pragma unroll
        for (int r = 0; r < 8; ++r) {
            int mrow = r + ((lane < 16) ? 0 : 8);
            float v = acc[nt][r] + bc;
            o2[(base + mrow) * NO + col] = gate ? sigmoidf_(v) : 0.0f;
        }
    }
}

// ---------------------------------------------------------------------------
// Order 1: tile = fixed b, i = it*16+m.  Branches bits 2/3/4.
// ---------------------------------------------------------------------------
__global__ void gemm1_kernel(const float* __restrict__ x0, const float* __restrict__ x1,
                             const float* __restrict__ redB,
                             const int* __restrict__ act,
                             const _Float16* __restrict__ wf_exp1,
                             const _Float16* __restrict__ wf_op1,
                             const _Float16* __restrict__ wf_red1,
                             const float* __restrict__ b_exp1,
                             const float* __restrict__ b_op1,
                             const float* __restrict__ b_red1,
                             float* __restrict__ o1) {
    int gid  = blockIdx.x * blockDim.x + threadIdx.x;
    int wave = gid >> 5;
    int lane = threadIdx.x & 31;
    int it = wave & 1;
    int b  = wave >> 1;
    int m  = lane & 15;
    int i  = it * 16 + m;

    int m2 = act[b * 10 + 2], m3 = act[b * 10 + 3], m4 = act[b * 10 + 4];
    int gate = 0;
#pragma unroll
    for (int bb = 0; bb < NB; ++bb)
        gate |= act[bb * 10 + 2] | act[bb * 10 + 3] | act[bb * 10 + 4];

    v8f z = {0.f, 0.f, 0.f, 0.f, 0.f, 0.f, 0.f, 0.f};
    v8f acc[4] = { z, z, z, z };

    if (m2) mm_chunk(acc, x0 + (size_t)b * NC, wf_exp1, 0, lane);          // expand(x0)
    if (m3) mm_chunk(acc, x1 + ((size_t)(b * NN + i)) * NC, wf_op1, 0, lane);
    if (m4) {
        const float* r = redB + ((size_t)(b * NN + i)) * (2 * NC);
        mm_chunk(acc, r,      wf_red1, 0, lane);
        mm_chunk(acc, r + NC, wf_red1, 1, lane);
    }

    size_t base = (size_t)(b * NN + it * 16);
    int ncol = lane & 15;
    float fm2 = (float)m2, fm3 = (float)m3, fm4 = (float)m4;
#pragma unroll
    for (int nt = 0; nt < 4; ++nt) {
        int col = nt * 16 + ncol;
        float bc = fm2 * b_exp1[col] + fm3 * b_op1[col] + fm4 * b_red1[col];
#pragma unroll
        for (int r = 0; r < 8; ++r) {
            int mrow = r + ((lane < 16) ? 0 : 8);
            float v = acc[nt][r] + bc;
            o1[(base + mrow) * NO + col] = gate ? sigmoidf_(v) : 0.0f;
        }
    }
}

// ---------------------------------------------------------------------------
// Order 0: single 16-row tile, rows 0..7 are the batches. Per-row masks are
// folded into the A operand (scale 0/1) so EXEC stays all-1s for WMMA.
// ---------------------------------------------------------------------------
__global__ void gemm0_kernel(const float* __restrict__ x0,
                             const float* __restrict__ redC,
                             const int* __restrict__ act,
                             const _Float16* __restrict__ wf_op0,
                             const _Float16* __restrict__ wf_red0,
                             const float* __restrict__ b_op0,
                             const float* __restrict__ b_red0,
                             float* __restrict__ o0) {
    int lane = threadIdx.x & 31;
    int m = lane & 15;
    int bm = (m < NB) ? m : (NB - 1);   // clamp loads for padded rows
    float sc0 = (m < NB) ? (float)act[bm * 10 + 0] : 0.0f;
    float sc1 = (m < NB) ? (float)act[bm * 10 + 1] : 0.0f;

    int gate = 0;
#pragma unroll
    for (int bb = 0; bb < NB; ++bb) gate |= act[bb * 10 + 0] | act[bb * 10 + 1];

    v8f z = {0.f, 0.f, 0.f, 0.f, 0.f, 0.f, 0.f, 0.f};
    v8f acc[4] = { z, z, z, z };

    mm_chunk_s(acc, x0 + (size_t)bm * NC, wf_op0, 0, lane, sc0);
    {
        const float* r = redC + (size_t)bm * (2 * NC);
        mm_chunk_s(acc, r,      wf_red0, 0, lane, sc1);
        mm_chunk_s(acc, r + NC, wf_red0, 1, lane, sc1);
    }

    if (lane < 16) {           // only D rows 0..7 (valid batches) are stored
        int ncol = lane;
#pragma unroll
        for (int nt = 0; nt < 4; ++nt) {
            int col = nt * 16 + ncol;
#pragma unroll
            for (int r = 0; r < NB; ++r) {   // rows 0..7
                float a0 = (float)act[r * 10 + 0];
                float a1 = (float)act[r * 10 + 1];
                float v = acc[nt][r] + a0 * b_op0[col] + a1 * b_red0[col];
                o0[(size_t)r * NO + col] = gate ? sigmoidf_(v) : 0.0f;
            }
        }
    }
}

// ---------------------------------------------------------------------------
// Host launcher
// ---------------------------------------------------------------------------
extern "C" void kernel_launch(void* const* d_in, const int* in_sizes, int n_in,
                              void* d_out, int out_size, void* d_ws, size_t ws_size,
                              hipStream_t stream) {
    const float* x0 = (const float*)d_in[0];
    const float* x1 = (const float*)d_in[1];
    const float* x2 = (const float*)d_in[2];
    const float* x3 = (const float*)d_in[3];
    const float* w_op0  = (const float*)d_in[4];   const float* b_op0  = (const float*)d_in[5];
    const float* w_red0 = (const float*)d_in[6];   const float* b_red0 = (const float*)d_in[7];
    const float* w_exp1 = (const float*)d_in[8];   const float* b_exp1 = (const float*)d_in[9];
    const float* w_op1  = (const float*)d_in[10];  const float* b_op1  = (const float*)d_in[11];
    const float* w_red1 = (const float*)d_in[12];  const float* b_red1 = (const float*)d_in[13];
    const float* w_exp2 = (const float*)d_in[14];  const float* b_exp2 = (const float*)d_in[15];
    const float* w_op2  = (const float*)d_in[16];  const float* b_op2  = (const float*)d_in[17];
    const float* w_red2 = (const float*)d_in[18];  const float* b_red2 = (const float*)d_in[19];
    const float* w_exp3 = (const float*)d_in[20];  const float* b_exp3 = (const float*)d_in[21];
    const float* w_op3  = (const float*)d_in[22];  const float* b_op3  = (const float*)d_in[23];
    const int*   act    = (const int*)d_in[24];

    float* out = (float*)d_out;
    float* o0 = out;
    float* o1 = out + NB * NO;                              // 512
    float* o2 = o1 + (size_t)NB * NN * NO;                  // +16384
    float* o3 = o2 + (size_t)NB * NN * NN * NO;             // +524288

    char* ws = (char*)d_ws;
    _Float16* wf   = (_Float16*)ws;
    float*    redA = (float*)(ws + WS_REDA_OFF);
    float*    redB = (float*)(ws + WS_REDB_OFF);
    float*    redC = (float*)(ws + WS_REDC_OFF);
    _Float16* x3h  = (_Float16*)(ws + WS_X3H_OFF);
    _Float16* x2h  = (_Float16*)(ws + WS_X2H_OFF);
    const bool f16_path = (ws_size >= (size_t)WS_F16_END);

    // --- weight prep (fp32 [O,K] -> f16 WMMA-B fragments) ---
    struct { const float* src; int off; int K; } wspec[10] = {
        { w_op0,  WOFF_OP0,  64 },  { w_red0, WOFF_RED0, 128 },
        { w_exp1, WOFF_EXP1, 64 },  { w_op1,  WOFF_OP1,  64 },
        { w_red1, WOFF_RED1, 128 }, { w_exp2, WOFF_EXP2, 128 },
        { w_op2,  WOFF_OP2,  128 }, { w_red2, WOFF_RED2, 256 },
        { w_exp3, WOFF_EXP3, 384 }, { w_op3,  WOFF_OP3,  384 },
    };
    for (int t = 0; t < 10; ++t) {
        int total = wspec[t].K * NO;
        int blocks = (total + 255) / 256;
        prep_w_kernel<<<blocks, 256, 0, stream>>>(wspec[t].src, wf + wspec[t].off,
                                                  wspec[t].K, total);
    }

    // --- reductions (single pass over x3 / x2 / x1) ---
    reduceA_kernel<<<(NB * NN * NN * NC) / 256, 256, 0, stream>>>(x3, redA);
    reduceB_kernel<<<(NB * NN * NC) / 256, 256, 0, stream>>>(x2, redB);
    reduceC_kernel<<<2, 256, 0, stream>>>(x1, redC);

    // --- WMMA GEMMs, heaviest first ---
    if (f16_path) {
        // one-time fp32->f16 conversion of gemm3 activations (x3h stays L2-resident)
        int t3 = NB * NN * NN * NN * NC / 4;    // 4,194,304 threads
        conv_f16_kernel<<<t3 / 256, 256, 0, stream>>>(x3, x3h, t3);
        int t2 = NB * NN * NN * NC / 4;         // 131,072 threads
        conv_f16_kernel<<<t2 / 256, 256, 0, stream>>>(x2, x2h, t2);
        gemm3_kernel<_Float16><<<2048, 256, 0, stream>>>(x2h, x3h, act,
                                                         wf + WOFF_EXP3, wf + WOFF_OP3,
                                                         b_exp3, b_op3, o3);
    } else {
        gemm3_kernel<float><<<2048, 256, 0, stream>>>(x2, x3, act,
                                                      wf + WOFF_EXP3, wf + WOFF_OP3,
                                                      b_exp3, b_op3, o3);
    }
    // order 2: 512 waves
    gemm2_kernel<<<64, 256, 0, stream>>>(x1, x2, redA, act,
                                         wf + WOFF_EXP2, wf + WOFF_OP2, wf + WOFF_RED2,
                                         b_exp2, b_op2, b_red2, o2);
    // order 1: 16 waves
    gemm1_kernel<<<2, 256, 0, stream>>>(x0, x1, redB, act,
                                        wf + WOFF_EXP1, wf + WOFF_OP1, wf + WOFF_RED1,
                                        b_exp1, b_op1, b_red1, o1);
    // order 0: 1 wave
    gemm0_kernel<<<1, 32, 0, stream>>>(x0, redC, act,
                                       wf + WOFF_OP0, wf + WOFF_RED0,
                                       b_op0, b_red0, o0);
    (void)in_sizes; (void)n_in; (void)out_size;
}
